// VectorQuantizer_566935683707
// MI455X (gfx1250) — matrix operand order repaired and checked by reading, not compiled, not run on hardware
//
#include <hip/hip_runtime.h>

// ---------------------------------------------------------------------------
// VQ-VAE vector quantizer for MI455X (gfx1250, wave32, WMMA).
//
// Core: 65536x8192 distance argmin via fused f16 WMMA GEMM (f32 accumulate),
// distances reduced to  score_k = ||e_k||^2 - 2 * z.e_k  (row-constant ||z||^2
// dropped). Codebook is pre-swizzled into WMMA B-fragment layout in ws.
// ---------------------------------------------------------------------------

typedef __attribute__((ext_vector_type(16))) _Float16 v16h;
typedef __attribute__((ext_vector_type(8)))  float    v8f;

#define N_TOK    65536     // 16*64*64 tokens
#define N_E      8192
#define E_DIM    64
#define HW       4096      // 64*64
#define CHW      262144    // 64*HW
#define ZQ_ELEMS 4194304   // 16*64*64*64
#define LOSS_OFF 4194304
#define IDX_OFF  4194305
#define BETA     0.25f

// ws byte layout:
//   [0, 1048576)          f16 B-fragment table: [tile 0..511][frag 0..1][lane 0..31][h 0..15]
//   [1048576, 1081344)    float e_norm[8192]
//   [1081344, 1343488)    int   idx[65536]
//   [1343488, 1409024)    float partial[16384]
#define WS_ENORM_OFF 1048576
#define WS_IDX_OFF   1081344
#define WS_PART_OFF  1343488

// ---------------------------------------------------------------------------
// Kernel 1: codebook -> f16 fragments (pre-swizzled) + squared norms.
// B (32x16) fragment k order this file uses for V_WMMA_F32_16X16X32_F16 (same as A):
//   lane = n + 16*hi,  half h -> inner K = frag*32 + 16*hi + h,  n = code-in-tile
// ---------------------------------------------------------------------------
__global__ void vq_prep(const float* __restrict__ emb,
                        _Float16* __restrict__ tbl,
                        float* __restrict__ enorm) {
    int k = blockIdx.x * blockDim.x + threadIdx.x;   // 0..8191
    int t = k >> 4;
    int n = k & 15;
    float s = 0.f;
    #pragma unroll
    for (int d = 0; d < E_DIM; ++d) {
        float v = emb[(size_t)k * E_DIM + d];
        s += v * v;
        int frag = d >> 5;
        int hi   = (d >> 4) & 1;
        int h    = d & 15;
        int lane = n + 16 * hi;
        tbl[((size_t)((t * 2 + frag) * 32 + lane)) * 16 + h] = (_Float16)v;
    }
    enorm[k] = s;
}

// ---------------------------------------------------------------------------
// Kernel 2: fused GEMM + argmin. 512 blocks x 256 threads (8 waves).
// Each wave owns one 16-row tile of z (A fragments loaded once, reused for all
// 8192 codes). 8 waves share B fragments staged through LDS (16 KB chunks of
// 8 code-tiles), cutting L2 traffic 8x. Inner loop: 2 wmma per 16-code tile.
// ---------------------------------------------------------------------------
__global__ __launch_bounds__(256) void vq_argmin(
        const float* __restrict__ z,
        const _Float16* __restrict__ tbl,
        const float* __restrict__ enorm,
        int* __restrict__ idx_out,
        float* __restrict__ idx_out_f) {
    __shared__ __align__(32) _Float16 sB[8 * 1024];   // 8 tiles * (2 frags*32 lanes*16 h)
    __shared__ float sN[128];                          // 8 tiles * 16 norms

    const int tid  = threadIdx.x;
    const int wave = tid >> 5;
    const int lane = tid & 31;
    const int m    = lane & 15;
    const int hi   = lane >> 4;
    const int ncol = m;

    const int rt     = blockIdx.x * 8 + wave;   // row-tile 0..4095
    const int n_base = rt * 16;

    // ---- load A fragments (16 rows x 64 dims, f32 -> f16), k order used here (same as B table):
    //   half h(0..15) -> inner K = fragBase + 16*hi + h
    const int n_row = n_base + m;
    const int b     = n_row >> 12;
    const int hw    = n_row & 4095;
    const float* zb = z + (size_t)b * CHW + hw;   // feature stride = HW floats
    v16h a0, a1;
    #pragma unroll
    for (int j = 0; j < 8; ++j) {
        a0[j]     = (_Float16)zb[(size_t)(16 * hi + j)       * HW];
        a0[j + 8] = (_Float16)zb[(size_t)(16 * hi + 8 + j)  * HW];
        a1[j]     = (_Float16)zb[(size_t)(32 + 16 * hi + j)  * HW];
        a1[j + 8] = (_Float16)zb[(size_t)(40 + 16 * hi + j)  * HW];
    }

    float minv[8];
    int   mini[8];
    #pragma unroll
    for (int r = 0; r < 8; ++r) { minv[r] = 3.4e38f; mini[r] = 0; }

    for (int tc = 0; tc < 64; ++tc) {            // 64 chunks of 8 code-tiles
        __syncthreads();
        {   // cooperative stage: 16 KB of B fragments + 128 norms
            const uint4* src = (const uint4*)(tbl + (size_t)tc * 8192);
            uint4* dst = (uint4*)sB;             // 1024 uint4 / 256 threads = 4 each
            #pragma unroll
            for (int j = 0; j < 4; ++j) dst[tid + 256 * j] = src[tid + 256 * j];
            if (tid < 128) sN[tid] = enorm[tc * 128 + tid];
        }
        __syncthreads();

        #pragma unroll
        for (int j = 0; j < 8; ++j) {
            v16h b0 = *(const v16h*)(sB + j * 1024 + lane * 16);
            v16h b1 = *(const v16h*)(sB + j * 1024 + 512 + lane * 16);
            float en = sN[j * 16 + ncol];
            v8f acc = {};
            acc = __builtin_amdgcn_wmma_f32_16x16x32_f16(
                      false, a0, false, b0, (short)0, acc, false, false);
            acc = __builtin_amdgcn_wmma_f32_16x16x32_f16(
                      false, a1, false, b1, (short)0, acc, false, false);
            const int k = (tc * 8 + j) * 16 + ncol;
            #pragma unroll
            for (int r = 0; r < 8; ++r) {        // score = ||e||^2 - 2*dot
                float s = __builtin_fmaf(-2.0f, acc[r], en);
                if (s < minv[r]) { minv[r] = s; mini[r] = k; }  // strict < keeps lowest k
            }
        }
    }

    // ---- cross-lane argmin over the 16 columns (n dimension); ties -> lower k
    #pragma unroll
    for (int mask = 8; mask >= 1; mask >>= 1) {
        #pragma unroll
        for (int r = 0; r < 8; ++r) {
            float ov = __shfl_xor(minv[r], mask, 32);
            int   oi = __shfl_xor(mini[r], mask, 32);
            if (ov < minv[r] || (ov == minv[r] && oi < mini[r])) {
                minv[r] = ov; mini[r] = oi;
            }
        }
    }
    if (ncol == 0) {   // lane 0 -> rows 0..7, lane 16 -> rows 8..15
        #pragma unroll
        for (int r = 0; r < 8; ++r) {
            int row = n_base + hi * 8 + r;
            idx_out[row]   = mini[r];
            idx_out_f[row] = (float)mini[r];
        }
    }
}

// ---------------------------------------------------------------------------
// Kernel 3: gather z_q (f32 codebook rows) into b,c,h,w layout + per-block
// squared-error partials. Fully coalesced: consecutive i -> consecutive w.
// ---------------------------------------------------------------------------
__global__ __launch_bounds__(256) void vq_gather(
        const float* __restrict__ z,
        const float* __restrict__ emb,
        const int* __restrict__ idx,
        float* __restrict__ zq,
        float* __restrict__ partial) {
    __shared__ float red[256];
    const int i  = blockIdx.x * 256 + threadIdx.x;   // flat b,c,h,w index
    const int b  = i >> 18;
    const int c  = (i >> 12) & 63;
    const int hw = i & 4095;
    const int n  = b * 4096 + hw;
    const int k  = idx[n];
    const float q = emb[(size_t)k * E_DIM + c];
    zq[i] = q;
    const float d = q - z[i];
    red[threadIdx.x] = d * d;
    __syncthreads();
    for (int s = 128; s > 0; s >>= 1) {
        if (threadIdx.x < s) red[threadIdx.x] += red[threadIdx.x + s];
        __syncthreads();
    }
    if (threadIdx.x == 0) partial[blockIdx.x] = red[0];
}

// ---------------------------------------------------------------------------
// Kernel 4: deterministic final reduction of 16384 partials -> loss scalar.
// loss = (1+beta) * mean((z_q - z)^2)   (stop_gradient is numerically a no-op)
// ---------------------------------------------------------------------------
__global__ __launch_bounds__(256) void vq_loss(
        const float* __restrict__ partial,
        float* __restrict__ loss) {
    __shared__ float red[256];
    float s = 0.f;
    for (int j = 0; j < 64; ++j) s += partial[threadIdx.x + 256 * j];
    red[threadIdx.x] = s;
    __syncthreads();
    for (int t = 128; t > 0; t >>= 1) {
        if (threadIdx.x < t) red[threadIdx.x] += red[threadIdx.x + t];
        __syncthreads();
    }
    if (threadIdx.x == 0)
        *loss = (1.0f + BETA) * red[0] / (float)ZQ_ELEMS;
}

// ---------------------------------------------------------------------------
extern "C" void kernel_launch(void* const* d_in, const int* in_sizes, int n_in,
                              void* d_out, int out_size, void* d_ws, size_t ws_size,
                              hipStream_t stream) {
    const float* z   = (const float*)d_in[0];   // (16,64,64,64) f32
    const float* emb = (const float*)d_in[1];   // (8192,64) f32
    float* out = (float*)d_out;                 // [z_q | loss | indices(as f32)]
    char* ws = (char*)d_ws;

    _Float16* tbl  = (_Float16*)ws;
    float* enorm   = (float*)(ws + WS_ENORM_OFF);
    int*   idx     = (int*)  (ws + WS_IDX_OFF);
    float* partial = (float*)(ws + WS_PART_OFF);

    vq_prep  <<<N_E / 256, 256, 0, stream>>>(emb, tbl, enorm);
    vq_argmin<<<N_TOK / 16 / 8, 256, 0, stream>>>(z, tbl, enorm, idx, out + IDX_OFF);
    vq_gather<<<ZQ_ELEMS / 256, 256, 0, stream>>>(z, emb, idx, out, partial);
    vq_loss  <<<1, 256, 0, stream>>>(partial, out + LOSS_OFF);
}
